// Decoder_49289044688964
// MI455X (gfx1250) — matrix operand order repaired and checked by reading, not compiled, not run on hardware
//
#include <hip/hip_runtime.h>
#include <hip/hip_bf16.h>

// ---------------------------------------------------------------------------
// Tacotron2-style decoder for MI455X (gfx1250, wave32, WMMA bf16x3 fp32-emul).
// Fast path: weights/state pre-split into bf16 hi/lo in WMMA fragment layout,
// so the GEMM hot loop is pure {b128 loads + v_wmma}.
// ---------------------------------------------------------------------------

typedef __attribute__((ext_vector_type(16))) __bf16 vbf16x16; // 8 VGPRs
typedef __attribute__((ext_vector_type(8)))  float  vf32x8;   // 8 VGPRs

#define BATCH   256
#define NMEL    80
#define TMEL    800
#define RSTEP   16
#define TSTEPS  50
#define RNN     1024
#define ENC     256
#define INDIM   1280      // RNN + ENC
#define MELD    1280
#define G4      4096      // 4*RNN
#define KTOT    2304      // INDIM + RNN

#define KT_G    (KTOT / 32)    // 72 k-tiles for gates GEMM
#define KT_X    (INDIM / 32)   // 40 k-tiles in x region
#define KT_H    (RNN / 32)     // 32 k-tiles in h region
#define NT_G    (G4 / 16)      // 256 n-tiles (gates)
#define NT_P    (MELD / 16)    // 80 n-tiles (mel proj)
#define MT      (BATCH / 16)   // 16 m-tiles

// packed fragment array sizes (elements, each of hi and lo)
#define WG_ELEMS ((size_t)NT_G * KT_G * 32 * 16)   // 9,437,184
#define WP_ELEMS ((size_t)NT_P * KT_H * 32 * 16)   // 1,310,720
#define X_ELEMS  ((size_t)MT * KT_X * 32 * 16)     //   327,680
#define H_ELEMS  ((size_t)MT * KT_H * 32 * 16)     //   262,144

#define FLOAT_WS ((size_t)(BATCH*G4 + 2*BATCH*RNN + BATCH*MELD + BATCH)) // 1,900,800 fl
#define BF_ELEMS (2 * (WG_ELEMS + WP_ELEMS + X_ELEMS + H_ELEMS))
#define WS_FAST_BYTES (FLOAT_WS * 4 + BF_ELEMS * 2)

static __device__ inline vf32x8 zero_v8() {
  vf32x8 z;
#pragma unroll
  for (int i = 0; i < 8; ++i) z[i] = 0.0f;
  return z;
}

static __device__ inline float sigf(float x) { return 1.0f / (1.0f + __expf(-x)); }

static __device__ inline void put_pair(vbf16x16& hi, vbf16x16& lo, int j, float a0, float a1) {
  __bf16 h0 = (__bf16)a0;
  __bf16 h1 = (__bf16)a1;
  hi[2 * j]     = h0;
  hi[2 * j + 1] = h1;
  lo[2 * j]     = (__bf16)(a0 - (float)h0);
  lo[2 * j + 1] = (__bf16)(a1 - (float)h1);
}

// Fetch one pre-packed fragment: lane's 16 bf16 live at a contiguous 32B slot.
static __device__ inline vbf16x16 ldfrag(const __bf16* __restrict__ base,
                                         int tile, int nkt, int kt, int lane) {
  return *(const vbf16x16*)(base + ((((size_t)tile * nkt + kt) * 32 + lane) << 4));
}

// bf16x3 emulated fp32 product: hi*hi + hi*lo + lo*hi, fp32 accumulate
static __device__ inline vf32x8 wmma3(vf32x8 acc, vbf16x16 ahi, vbf16x16 alo,
                                      vbf16x16 bhi, vbf16x16 blo) {
  acc = __builtin_amdgcn_wmma_f32_16x16x32_bf16(false, ahi, false, bhi, (short)0, acc, false, false);
  acc = __builtin_amdgcn_wmma_f32_16x16x32_bf16(false, ahi, false, blo, (short)0, acc, false, false);
  acc = __builtin_amdgcn_wmma_f32_16x16x32_bf16(false, alo, false, bhi, (short)0, acc, false, false);
  return acc;
}

// ---------------------------------------------------------------------------
// Packing kernels (fragment layouts per CDNA5 ISA §7.12.2)
// B fragment: lane = khalf*16 + n%16 ; elem e (=2j+p) -> K = kt*32 + (j/4)*16 + khalf*8 + (j%4)*2 + p
// A fragment: lane = half*16 + m%16 ; elem 2j+p -> K = kt*32 + (j/4)*16 + half*8 + (j%4)*2 + p
// ---------------------------------------------------------------------------
__global__ void pack_wg(const float* __restrict__ W_ih, const float* __restrict__ W_hh,
                        __bf16* __restrict__ hi, __bf16* __restrict__ lo) {
  int idx = blockIdx.x * blockDim.x + threadIdx.x;   // NT_G*KT_G*32 threads
  if (idx >= NT_G * KT_G * 32) return;
  int lane = idx & 31;
  int rest = idx >> 5;
  int kt = rest % KT_G;
  int ntile = rest / KT_G;
  int n = ntile * 16 + (lane & 15);
  int kbase = kt * 32 + ((lane >> 4) << 3);
  __bf16* ph = hi + (size_t)idx * 16;
  __bf16* pl = lo + (size_t)idx * 16;
#pragma unroll
  for (int e = 0; e < 16; ++e) {
    int k = kbase + (e & 7) + ((e >> 3) << 4);
    float v = (k < INDIM) ? W_ih[(size_t)n * INDIM + k]
                          : W_hh[(size_t)n * RNN + (k - INDIM)];
    __bf16 vh = (__bf16)v;
    ph[e] = vh;
    pl[e] = (__bf16)(v - (float)vh);
  }
}

__global__ void pack_wp(const float* __restrict__ W_proj,
                        __bf16* __restrict__ hi, __bf16* __restrict__ lo) {
  int idx = blockIdx.x * blockDim.x + threadIdx.x;   // NT_P*KT_H*32 threads
  if (idx >= NT_P * KT_H * 32) return;
  int lane = idx & 31;
  int rest = idx >> 5;
  int kt = rest % KT_H;
  int ntile = rest / KT_H;
  int n = ntile * 16 + (lane & 15);
  int kbase = kt * 32 + ((lane >> 4) << 3);
  const float* row = W_proj + (size_t)n * INDIM + kbase;   // k < 1024
  __bf16* ph = hi + (size_t)idx * 16;
  __bf16* pl = lo + (size_t)idx * 16;
#pragma unroll
  for (int e = 0; e < 16; ++e) {
    float v = row[(e & 7) + ((e >> 3) << 4)];
    __bf16 vh = (__bf16)v;
    ph[e] = vh;
    pl[e] = (__bf16)(v - (float)vh);
  }
}

// x_t[b][d] = target[b][d%80][t*16 + d/80]  packed as A fragments
__global__ void pack_x(const float* __restrict__ target,
                       __bf16* __restrict__ hi, __bf16* __restrict__ lo, int t) {
  int idx = blockIdx.x * blockDim.x + threadIdx.x;   // MT*KT_X*32 threads
  if (idx >= MT * KT_X * 32) return;
  int lane = idx & 31;
  int rest = idx >> 5;
  int kt = rest % KT_X;
  int mtile = rest / KT_X;
  int m = mtile * 16 + (lane & 15);
  int half = lane >> 4;
  const float* base = target + (size_t)m * (NMEL * TMEL) + t * RSTEP;
  vbf16x16 vh, vl;
#pragma unroll
  for (int j = 0; j < 8; ++j) {
    unsigned k = (unsigned)(kt * 32) + ((j >> 2) << 4) + (half << 3) + ((j & 3) << 1);
    unsigned jj = k / NMEL;
    unsigned mel = k - jj * NMEL;                    // even
    float a0 = base[(size_t)mel * TMEL + jj];
    float a1 = base[(size_t)(mel + 1) * TMEL + jj];
    put_pair(vh, vl, j, a0, a1);
  }
  *(vbf16x16*)(hi + (size_t)idx * 16) = vh;
  *(vbf16x16*)(lo + (size_t)idx * 16) = vl;
}

// ---------------------------------------------------------------------------
// FAST step kernel 1: gates = [x_t | h] @ [W_ih | W_hh]^T  (pre-packed operands)
// grid (32, 4), block 256 (8 waves, 2(M) x 4(N)), wave tile 32x32
// ---------------------------------------------------------------------------
__global__ __launch_bounds__(256) void gates_fast(
    const __bf16* __restrict__ Xhi, const __bf16* __restrict__ Xlo,
    const __bf16* __restrict__ Hhi, const __bf16* __restrict__ Hlo,
    const __bf16* __restrict__ Wghi, const __bf16* __restrict__ Wglo,
    float* __restrict__ gates, int t) {
  int wave = threadIdx.x >> 5;
  int lane = threadIdx.x & 31;
  int half = lane >> 4;
  int lhs  = lane & 15;
  int wm = wave & 1, wn = wave >> 1;
  int mt0 = blockIdx.y * 4 + wm * 2;
  int nt0 = blockIdx.x * 8 + wn * 2;

  vf32x8 acc[2][2];
#pragma unroll
  for (int a = 0; a < 2; ++a)
#pragma unroll
    for (int b = 0; b < 2; ++b) acc[a][b] = zero_v8();

  if (t != 0) {   // go-frame is all zeros: skip x region at t==0
#pragma unroll 2
    for (int kt = 0; kt < KT_X; ++kt) {
      vbf16x16 a0h = ldfrag(Xhi, mt0,     KT_X, kt, lane);
      vbf16x16 a0l = ldfrag(Xlo, mt0,     KT_X, kt, lane);
      vbf16x16 a1h = ldfrag(Xhi, mt0 + 1, KT_X, kt, lane);
      vbf16x16 a1l = ldfrag(Xlo, mt0 + 1, KT_X, kt, lane);
#pragma unroll
      for (int tn = 0; tn < 2; ++tn) {
        vbf16x16 bh = ldfrag(Wghi, nt0 + tn, KT_G, kt, lane);
        vbf16x16 bl = ldfrag(Wglo, nt0 + tn, KT_G, kt, lane);
        acc[0][tn] = wmma3(acc[0][tn], a0h, a0l, bh, bl);
        acc[1][tn] = wmma3(acc[1][tn], a1h, a1l, bh, bl);
      }
    }
  }
#pragma unroll 2
  for (int kt = 0; kt < KT_H; ++kt) {
    vbf16x16 a0h = ldfrag(Hhi, mt0,     KT_H, kt, lane);
    vbf16x16 a0l = ldfrag(Hlo, mt0,     KT_H, kt, lane);
    vbf16x16 a1h = ldfrag(Hhi, mt0 + 1, KT_H, kt, lane);
    vbf16x16 a1l = ldfrag(Hlo, mt0 + 1, KT_H, kt, lane);
#pragma unroll
    for (int tn = 0; tn < 2; ++tn) {
      vbf16x16 bh = ldfrag(Wghi, nt0 + tn, KT_G, kt + KT_X, lane);
      vbf16x16 bl = ldfrag(Wglo, nt0 + tn, KT_G, kt + KT_X, lane);
      acc[0][tn] = wmma3(acc[0][tn], a0h, a0l, bh, bl);
      acc[1][tn] = wmma3(acc[1][tn], a1h, a1l, bh, bl);
    }
  }

#pragma unroll
  for (int tm = 0; tm < 2; ++tm)
#pragma unroll
    for (int tn = 0; tn < 2; ++tn) {
      int n = (nt0 + tn) * 16 + lhs;
#pragma unroll
      for (int r = 0; r < 8; ++r) {
        int m = (mt0 + tm) * 16 + r + (half << 3);
        gates[(size_t)m * G4 + n] = acc[tm][tn][r];
      }
    }
}

// ---------------------------------------------------------------------------
// FAST step kernel 3: mel = h @ W_proj[:, :1024]^T + Cctx, scattered to output
// grid (10, 4), block 256
// ---------------------------------------------------------------------------
__global__ __launch_bounds__(256) void mel_fast(
    const __bf16* __restrict__ Hhi, const __bf16* __restrict__ Hlo,
    const __bf16* __restrict__ Wphi, const __bf16* __restrict__ Wplo,
    const float* __restrict__ Cctx, float* __restrict__ out_mel, int t) {
  int wave = threadIdx.x >> 5;
  int lane = threadIdx.x & 31;
  int half = lane >> 4;
  int lhs  = lane & 15;
  int wm = wave & 1, wn = wave >> 1;
  int mt0 = blockIdx.y * 4 + wm * 2;
  int nt0 = blockIdx.x * 8 + wn * 2;

  vf32x8 acc[2][2];
#pragma unroll
  for (int tm = 0; tm < 2; ++tm)
#pragma unroll
    for (int tn = 0; tn < 2; ++tn) {
      int n = (nt0 + tn) * 16 + lhs;
#pragma unroll
      for (int r = 0; r < 8; ++r) {
        int m = (mt0 + tm) * 16 + r + (half << 3);
        acc[tm][tn][r] = Cctx[(size_t)m * MELD + n];   // context part + b_proj
      }
    }

#pragma unroll 2
  for (int kt = 0; kt < KT_H; ++kt) {
    vbf16x16 a0h = ldfrag(Hhi, mt0,     KT_H, kt, lane);
    vbf16x16 a0l = ldfrag(Hlo, mt0,     KT_H, kt, lane);
    vbf16x16 a1h = ldfrag(Hhi, mt0 + 1, KT_H, kt, lane);
    vbf16x16 a1l = ldfrag(Hlo, mt0 + 1, KT_H, kt, lane);
#pragma unroll
    for (int tn = 0; tn < 2; ++tn) {
      vbf16x16 bh = ldfrag(Wphi, nt0 + tn, KT_H, kt, lane);
      vbf16x16 bl = ldfrag(Wplo, nt0 + tn, KT_H, kt, lane);
      acc[0][tn] = wmma3(acc[0][tn], a0h, a0l, bh, bl);
      acc[1][tn] = wmma3(acc[1][tn], a1h, a1l, bh, bl);
    }
  }

#pragma unroll
  for (int tm = 0; tm < 2; ++tm)
#pragma unroll
    for (int tn = 0; tn < 2; ++tn) {
      unsigned n = (unsigned)((nt0 + tn) * 16 + lhs);
      unsigned jj = n / NMEL;
      unsigned mel = n - jj * NMEL;
#pragma unroll
      for (int r = 0; r < 8; ++r) {
        int m = (mt0 + tm) * 16 + r + (half << 3);
        out_mel[(size_t)m * (NMEL * TMEL) + (size_t)mel * TMEL + t * RSTEP + jj] =
            acc[tm][tn][r];
      }
    }
}

// ---------------------------------------------------------------------------
// Step kernel 2: LSTM pointwise; optionally emits packed bf16 A-fragments of h
// ---------------------------------------------------------------------------
__global__ void lstm_pointwise(const float* __restrict__ gates,
                               const float* __restrict__ b_ih,
                               const float* __restrict__ b_hh,
                               float* __restrict__ h, float* __restrict__ c,
                               __bf16* __restrict__ Hhi, __bf16* __restrict__ Hlo) {
  int idx = blockIdx.x * blockDim.x + threadIdx.x;
  if (idx >= BATCH * RNN) return;
  int b = idx >> 10, u = idx & (RNN - 1);
  const float* g = gates + (size_t)b * G4;
  float gi = g[u]           + b_ih[u]           + b_hh[u];
  float gf = g[u + RNN]     + b_ih[u + RNN]     + b_hh[u + RNN];
  float gg = g[u + 2 * RNN] + b_ih[u + 2 * RNN] + b_hh[u + 2 * RNN];
  float go = g[u + 3 * RNN] + b_ih[u + 3 * RNN] + b_hh[u + 3 * RNN];
  float cn = sigf(gf) * c[idx] + sigf(gi) * tanhf(gg);
  float hn = sigf(go) * tanhf(cn);
  c[idx] = cn;
  h[idx] = hn;
  if (Hhi) {  // scatter into packed A-fragment layout for next step's WMMAs
    int mtile = b >> 4, lhs = b & 15;
    int kt = u >> 5, kl = u & 31;
    int grp = kl >> 4, r = kl & 15;
    int half = r >> 3, r2 = r & 7;
    int j = grp * 4 + (r2 >> 1), p = r2 & 1;
    int lane = half * 16 + lhs;
    size_t addr = ((((size_t)mtile * KT_H + kt) * 32) + lane) * 16 + 2 * j + p;
    __bf16 vh = (__bf16)hn;
    Hhi[addr] = vh;
    Hlo[addr] = (__bf16)(hn - (float)vh);
  }
}

// ---------------------------------------------------------------------------
// Step kernel 4: gate logit per batch row (wave-per-row reduction)
// ---------------------------------------------------------------------------
__global__ void gate_kernel(const float* __restrict__ h,
                            const float* __restrict__ W_gate,
                            const float* __restrict__ gctx,
                            float* __restrict__ out_gate, int t) {
  int wave = threadIdx.x >> 5;
  int lane = threadIdx.x & 31;
  int b = blockIdx.x * 8 + wave;
  float s = 0.0f;
  const float* row = h + (size_t)b * RNN;
  for (int k = lane; k < RNN; k += 32) s += row[k] * W_gate[k];
#pragma unroll
  for (int off = 16; off > 0; off >>= 1) s += __shfl_down(s, off, 32);
  if (lane == 0) out_gate[(size_t)b * TSTEPS + t] = s + gctx[b];
}

// ---------------------------------------------------------------------------
// Prologue kernels
// ---------------------------------------------------------------------------
__global__ void init_state(float* __restrict__ h, float* __restrict__ c,
                           __bf16* __restrict__ Hhi, __bf16* __restrict__ Hlo) {
  int idx = blockIdx.x * blockDim.x + threadIdx.x;
  if (idx < BATCH * RNN) {
    h[idx] = 0.0f;
    c[idx] = 0.0f;
    if (Hhi) { Hhi[idx] = (__bf16)0.0f; Hlo[idx] = (__bf16)0.0f; }  // H_ELEMS == BATCH*RNN
  }
}

__global__ void ctx_proj(const float* __restrict__ context,
                         const float* __restrict__ W_proj,
                         const float* __restrict__ b_proj,
                         float* __restrict__ Cctx) {
  int idx = blockIdx.x * blockDim.x + threadIdx.x;
  if (idx >= BATCH * MELD) return;
  int b = idx / MELD, d = idx - b * MELD;
  float s = b_proj[d];
  const float* wr = W_proj + (size_t)d * INDIM + RNN;
  const float* cr = context + (size_t)b * ENC;
  for (int e = 0; e < ENC; ++e) s += cr[e] * wr[e];
  Cctx[idx] = s;
}

__global__ void ctx_gate(const float* __restrict__ context,
                         const float* __restrict__ W_gate,
                         const float* __restrict__ b_gate,
                         float* __restrict__ gctx) {
  int b = blockIdx.x * blockDim.x + threadIdx.x;
  if (b >= BATCH) return;
  float s = b_gate[0];
  const float* cr = context + (size_t)b * ENC;
  for (int e = 0; e < ENC; ++e) s += cr[e] * W_gate[RNN + e];
  gctx[b] = s;
}

// ---------------------------------------------------------------------------
// FALLBACK on-the-fly GEMM kernels (used when ws_size can't hold the packs)
// ---------------------------------------------------------------------------
static __device__ inline void loadA_rowmajor(const float* __restrict__ src, int ld,
                                             int m, int half, int k0,
                                             vbf16x16& hi, vbf16x16& lo) {
  const float* row = src + (size_t)m * ld + k0;
#pragma unroll
  for (int j = 0; j < 8; ++j) {
    int kl = ((j >> 2) << 4) + (half << 3) + ((j & 3) << 1);
    put_pair(hi, lo, j, row[kl], row[kl + 1]);
  }
}

static __device__ inline void loadA_target(const float* __restrict__ target,
                                           int b, int t, int half, int k0,
                                           vbf16x16& hi, vbf16x16& lo) {
  const float* base = target + (size_t)b * (NMEL * TMEL) + t * RSTEP;
#pragma unroll
  for (int j = 0; j < 8; ++j) {
    unsigned kl = ((j >> 2) << 4) + (half << 3) + ((j & 3) << 1);
    unsigned k  = (unsigned)k0 + kl;
    unsigned jj = k / NMEL;
    unsigned mel = k - jj * NMEL;
    float a0 = base[(size_t)mel * TMEL + jj];
    float a1 = base[(size_t)(mel + 1) * TMEL + jj];
    put_pair(hi, lo, j, a0, a1);
  }
}

static __device__ inline void loadB_w(const float* __restrict__ w, int ld,
                                      int n, int khalf, int k0,
                                      vbf16x16& hi, vbf16x16& lo) {
  const float* row = w + (size_t)n * ld + k0 + (khalf << 3);
#pragma unroll
  for (int j = 0; j < 8; ++j) {
    int kl = ((j >> 2) << 4) + ((j & 3) << 1);
    put_pair(hi, lo, j, row[kl], row[kl + 1]);
  }
}

__global__ __launch_bounds__(256) void gates_fb(
    const float* __restrict__ target, const float* __restrict__ h,
    const float* __restrict__ W_ih, const float* __restrict__ W_hh,
    float* __restrict__ gates, int t) {
  int wave = threadIdx.x >> 5, lane = threadIdx.x & 31;
  int half = lane >> 4, lhs = lane & 15;
  int wm = wave & 1, wn = wave >> 1;
  int m0 = blockIdx.y * 64 + wm * 32;
  int n0 = blockIdx.x * 128 + wn * 32;
  vf32x8 acc[2][2];
#pragma unroll
  for (int a = 0; a < 2; ++a)
#pragma unroll
    for (int b = 0; b < 2; ++b) acc[a][b] = zero_v8();
  const int ktStart = (t == 0) ? KT_X : 0;
  for (int kt = ktStart; kt < KT_G; ++kt) {
    int k0 = kt * 32;
    vbf16x16 ahi[2], alo[2];
    if (k0 < INDIM) {
      loadA_target(target, m0 + lhs,      t, half, k0, ahi[0], alo[0]);
      loadA_target(target, m0 + 16 + lhs, t, half, k0, ahi[1], alo[1]);
    } else {
      loadA_rowmajor(h, RNN, m0 + lhs,      half, k0 - INDIM, ahi[0], alo[0]);
      loadA_rowmajor(h, RNN, m0 + 16 + lhs, half, k0 - INDIM, ahi[1], alo[1]);
    }
#pragma unroll
    for (int tn = 0; tn < 2; ++tn) {
      vbf16x16 bhi, blo;
      int n = n0 + tn * 16 + lhs;
      if (k0 < INDIM) loadB_w(W_ih, INDIM, n, half, k0,         bhi, blo);
      else            loadB_w(W_hh, RNN,   n, half, k0 - INDIM, bhi, blo);
      acc[0][tn] = wmma3(acc[0][tn], ahi[0], alo[0], bhi, blo);
      acc[1][tn] = wmma3(acc[1][tn], ahi[1], alo[1], bhi, blo);
    }
  }
#pragma unroll
  for (int tm = 0; tm < 2; ++tm)
#pragma unroll
    for (int tn = 0; tn < 2; ++tn) {
      int n = n0 + tn * 16 + lhs;
#pragma unroll
      for (int r = 0; r < 8; ++r)
        gates[(size_t)(m0 + tm * 16 + r + (half << 3)) * G4 + n] = acc[tm][tn][r];
    }
}

__global__ __launch_bounds__(256) void mel_fb(
    const float* __restrict__ h, const float* __restrict__ W_proj,
    const float* __restrict__ Cctx, float* __restrict__ out_mel, int t) {
  int wave = threadIdx.x >> 5, lane = threadIdx.x & 31;
  int half = lane >> 4, lhs = lane & 15;
  int wm = wave & 1, wn = wave >> 1;
  int m0 = blockIdx.y * 64 + wm * 32;
  int n0 = blockIdx.x * 128 + wn * 32;
  vf32x8 acc[2][2];
#pragma unroll
  for (int tm = 0; tm < 2; ++tm)
#pragma unroll
    for (int tn = 0; tn < 2; ++tn) {
      int n = n0 + tn * 16 + lhs;
#pragma unroll
      for (int r = 0; r < 8; ++r)
        acc[tm][tn][r] = Cctx[(size_t)(m0 + tm * 16 + r + (half << 3)) * MELD + n];
    }
  for (int kt = 0; kt < KT_H; ++kt) {
    int k0 = kt * 32;
    vbf16x16 ahi[2], alo[2];
    loadA_rowmajor(h, RNN, m0 + lhs,      half, k0, ahi[0], alo[0]);
    loadA_rowmajor(h, RNN, m0 + 16 + lhs, half, k0, ahi[1], alo[1]);
#pragma unroll
    for (int tn = 0; tn < 2; ++tn) {
      vbf16x16 bhi, blo;
      int n = n0 + tn * 16 + lhs;
      loadB_w(W_proj, INDIM, n, half, k0, bhi, blo);
      acc[0][tn] = wmma3(acc[0][tn], ahi[0], alo[0], bhi, blo);
      acc[1][tn] = wmma3(acc[1][tn], ahi[1], alo[1], bhi, blo);
    }
  }
#pragma unroll
  for (int tm = 0; tm < 2; ++tm)
#pragma unroll
    for (int tn = 0; tn < 2; ++tn) {
      unsigned n = (unsigned)(n0 + tn * 16 + lhs);
      unsigned jj = n / NMEL;
      unsigned mel = n - jj * NMEL;
#pragma unroll
      for (int r = 0; r < 8; ++r)
        out_mel[(size_t)(m0 + tm * 16 + r + (half << 3)) * (NMEL * TMEL) +
                (size_t)mel * TMEL + t * RSTEP + jj] = acc[tm][tn][r];
    }
}

// ---------------------------------------------------------------------------
extern "C" void kernel_launch(void* const* d_in, const int* in_sizes, int n_in,
                              void* d_out, int out_size, void* d_ws, size_t ws_size,
                              hipStream_t stream) {
  const float* context = (const float*)d_in[0];
  const float* target  = (const float*)d_in[1];
  const float* W_ih    = (const float*)d_in[2];
  const float* b_ih    = (const float*)d_in[3];
  const float* W_hh    = (const float*)d_in[4];
  const float* b_hh    = (const float*)d_in[5];
  const float* W_proj  = (const float*)d_in[6];
  const float* b_proj  = (const float*)d_in[7];
  const float* W_gate  = (const float*)d_in[8];
  const float* b_gate  = (const float*)d_in[9];

  float* out_mel  = (float*)d_out;                         // 256*80*800
  float* out_gate = out_mel + (size_t)BATCH * NMEL * TMEL; // 256*50

  float* ws    = (float*)d_ws;
  float* gates = ws;                                   // 256*4096
  float* h     = gates + (size_t)BATCH * G4;
  float* c     = h + (size_t)BATCH * RNN;
  float* Cctx  = c + (size_t)BATCH * RNN;
  float* gctx  = Cctx + (size_t)BATCH * MELD;

  const bool fast = ws_size >= WS_FAST_BYTES;
  __bf16* bfbase = (__bf16*)(ws + FLOAT_WS);           // 32B-aligned (offset % 32 == 0)
  __bf16* Wghi = bfbase;
  __bf16* Wglo = Wghi + WG_ELEMS;
  __bf16* Wphi = Wglo + WG_ELEMS;
  __bf16* Wplo = Wphi + WP_ELEMS;
  __bf16* Xhi  = Wplo + WP_ELEMS;
  __bf16* Xlo  = Xhi + X_ELEMS;
  __bf16* Hhi  = Xlo + X_ELEMS;
  __bf16* Hlo  = Hhi + H_ELEMS;

  init_state<<<(BATCH * RNN + 255) / 256, 256, 0, stream>>>(
      h, c, fast ? Hhi : (__bf16*)nullptr, fast ? Hlo : (__bf16*)nullptr);
  ctx_proj<<<(BATCH * MELD + 255) / 256, 256, 0, stream>>>(context, W_proj, b_proj, Cctx);
  ctx_gate<<<1, 256, 0, stream>>>(context, W_gate, b_gate, gctx);

  if (fast) {
    pack_wg<<<(NT_G * KT_G * 32 + 255) / 256, 256, 0, stream>>>(W_ih, W_hh, Wghi, Wglo);
    pack_wp<<<(NT_P * KT_H * 32 + 255) / 256, 256, 0, stream>>>(W_proj, Wphi, Wplo);
  }

  dim3 gGrid(G4 / 128, BATCH / 64);   // (32, 4)
  dim3 mGrid(MELD / 128, BATCH / 64); // (10, 4)
  for (int t = 0; t < TSTEPS; ++t) {
    if (fast) {
      if (t != 0)
        pack_x<<<(MT * KT_X * 32 + 255) / 256, 256, 0, stream>>>(target, Xhi, Xlo, t);
      gates_fast<<<gGrid, 256, 0, stream>>>(Xhi, Xlo, Hhi, Hlo, Wghi, Wglo, gates, t);
      lstm_pointwise<<<(BATCH * RNN + 255) / 256, 256, 0, stream>>>(
          gates, b_ih, b_hh, h, c, Hhi, Hlo);
      mel_fast<<<mGrid, 256, 0, stream>>>(Hhi, Hlo, Wphi, Wplo, Cctx, out_mel, t);
    } else {
      gates_fb<<<gGrid, 256, 0, stream>>>(target, h, W_ih, W_hh, gates, t);
      lstm_pointwise<<<(BATCH * RNN + 255) / 256, 256, 0, stream>>>(
          gates, b_ih, b_hh, h, c, (__bf16*)nullptr, (__bf16*)nullptr);
      mel_fb<<<mGrid, 256, 0, stream>>>(h, W_proj, Cctx, out_mel, t);
    }
    gate_kernel<<<BATCH / 8, 256, 0, stream>>>(h, W_gate, gctx, out_gate, t);
  }
}
